// SCSSystem_3796751089945
// MI455X (gfx1250) — matrix operand order repaired and checked by reading, not compiled, not run on hardware
//
#include <hip/hip_runtime.h>
#include <hip/hip_bf16.h>

typedef __attribute__((ext_vector_type(16))) _Float16 v16h;
typedef __attribute__((ext_vector_type(8)))  _Float16 v8h;
typedef __attribute__((ext_vector_type(8)))  float    v8f;
typedef unsigned int u32x4 __attribute__((ext_vector_type(4)));
typedef int          i32x4 __attribute__((ext_vector_type(4)));
typedef int          i32x8 __attribute__((ext_vector_type(8)));

#define Nn  32      // M*L nodes
#define Bb  16      // batch
#define HWp 16384   // 128*128 pixels
#define Ee  256     // connections
#define PIX 8       // pixels per workgroup

union AFrag { v16h v; v8h h[2]; };
union CFrag { v8f  v; float f[8]; };

__global__ __launch_bounds__(256) void scs_wmma_kernel(
    const float* __restrict__ spikes,       // [N,B,H,W]
    const float* __restrict__ exc_mask,     // [N,H,W]
    const float* __restrict__ conn_weight,  // [E,H,W]
    const float* __restrict__ scale_w,      // [2]
    const int*   __restrict__ conn_src,     // [E]
    const int*   __restrict__ conn_dst,     // [E]
    float*       __restrict__ out)          // [N,B,H,W]
{
    // 59 KB static LDS total
    __shared__ __align__(16) float    Wlds[PIX*Nn*Nn];  // 32 KB; reused as C staging
    __shared__ __align__(16) _Float16 A16 [PIX*Nn*Nn];  // 16 KB; front 8 KB = TDM landing pad
    __shared__ __align__(16) _Float16 S16 [PIX*Bb*Nn];  //  8 KB (spikes, f16, B-layout)
    __shared__ __align__(16) float    excl[Nn*PIX];     //  1 KB
    __shared__ int csrc[Ee];                            //  1 KB
    __shared__ int cdst[Ee];                            //  1 KB

    const int t    = threadIdx.x;
    const int pix0 = blockIdx.x * PIX;
    float* cwst = (float*)A16;  // 2048-float conn_weight tile, alive until phase 2

    // ---- Phase 0a: TDM — DMA the [E x PIX] f32 conn_weight tile into LDS ----
    // 2D tile: 256 rows x 8 elems, row stride HW elems, contiguous in LDS.
    if (t < 32) {
        const unsigned long long ga = (unsigned long long)(uintptr_t)(conn_weight + pix0);
        const unsigned int lds_addr = (unsigned int)(uintptr_t)(void*)cwst; // low32 = LDS byte offset
        u32x4 g0;
        g0[0] = 1u;                                   // count=1, user descriptor
        g0[1] = lds_addr;                             // lds_addr (bytes)
        g0[2] = (unsigned int)ga;                     // global_addr[31:0]
        g0[3] = (unsigned int)(ga >> 32) | (2u << 30);// global_addr[56:32] | type=2
        i32x8 g1;
        g1[0] = 0x00020000;                           // workgroup_mask=0, data_size=4B
        g1[1] = (int)((HWp & 0xFFFF) << 16);          // tensor_dim0[15:0]
        g1[2] = (int)((HWp >> 16) & 0xFFFF) | (int)((Ee & 0xFFFF) << 16); // dim0 hi | dim1 lo
        g1[3] = (int)((Ee >> 16) & 0xFFFF) | (PIX << 16);                 // dim1 hi | tile_dim0=8
        g1[4] = Ee;                                   // tile_dim1=256 (tile_dim2=0)
        g1[5] = HWp;                                  // tensor_dim0_stride[31:0]
        g1[6] = 0;                                    // stride hi, dim1_stride lo
        g1[7] = 0;
        i32x4 gz4 = {0, 0, 0, 0};                     // 2D: groups 2/3 unused
        i32x8 gz8 = {0, 0, 0, 0, 0, 0, 0, 0};        // extra group (clang-23 6-arg form)
        __builtin_amdgcn_tensor_load_to_lds(g0, g1, gz4, gz4, gz8, 0);
        __builtin_amdgcn_s_wait_tensorcnt(0);
    }

    // ---- Phase 0b: zero W, stage indices / exc / f16 spikes (all b128 global) ----
    {
        float4* W4 = (float4*)Wlds;
        #pragma unroll
        for (int k = 0; k < (PIX*Nn*Nn/4)/256; ++k)
            W4[t + k*256] = make_float4(0.f, 0.f, 0.f, 0.f);
    }
    csrc[t] = conn_src[t];
    cdst[t] = conn_dst[t];
    if (t < 64) {                                     // exc_mask: 64 x float4
        int p4 = t & 1, n2 = t >> 1;
        float4 e4 = *(const float4*)(exc_mask + n2*HWp + pix0 + p4*4);
        ((float4*)excl)[n2*2 + p4] = e4;              // excl[n2*8 + p]
    }
    #pragma unroll
    for (int k = 0; k < (PIX*Bb*Nn/4)/256; ++k) {     // spikes: float4 loads -> f16 B-layout
        int idx = t + k*256;
        int p4 = idx & 1, rest = idx >> 1;
        int b = rest & 15, n2 = rest >> 4;
        const float* sp = spikes + (n2*Bb + b)*HWp + pix0 + p4*4;
        __builtin_prefetch(sp + HWp, 0, 1);           // global_prefetch_b8
        float4 s4 = *(const float4*)sp;
        int pb = p4*4;
        S16[(pb+0)*(Bb*Nn) + b*Nn + n2] = (_Float16)s4.x;
        S16[(pb+1)*(Bb*Nn) + b*Nn + n2] = (_Float16)s4.y;
        S16[(pb+2)*(Bb*Nn) + b*Nn + n2] = (_Float16)s4.z;
        S16[(pb+3)*(Bb*Nn) + b*Nn + n2] = (_Float16)s4.w;
    }
    __syncthreads();

    // ---- Phase 1: assemble W_pix[dst,src] (deterministic: thread owns (pixel,dst) row) ----
    {   int p = t & 7, d = t >> 3;                    // d in [0,32)
        float* wrow = &Wlds[(p*Nn + d)*Nn];
        for (int e = 0; e < Ee; ++e) {
            if (cdst[e] == d) wrow[csrc[e]] += cwst[e*PIX + p];
        }
    }
    __syncthreads();

    // ---- Phase 2: A = W * exc_mask + G ; convert to f16 (row-major 32x32 per pixel) ----
    const float sw0 = 0.25f * scale_w[0];             // /L, spacing 1
    const float sw1 = 0.25f * scale_w[1];             // /L, spacing 2
    #pragma unroll
    for (int k = 0; k < (PIX*Nn*Nn)/256; ++k) {
        int idx = t + k*256;
        int p = idx >> 10, rem = idx & 1023;
        int n = rem >> 5, n2 = rem & 31;
        int dm = (n >> 2) - (n2 >> 2);
        int ad = dm < 0 ? -dm : dm;
        float g = (ad == 1) ? sw0 : ((ad == 2) ? sw1 : 0.0f);
        A16[idx] = (_Float16)(Wlds[idx] * excl[n2*PIX + p] + g);
    }
    __syncthreads();

    // ---- Phase 3: per-pixel (32x32)@(32x16) via two v_wmma_f32_16x16x32_f16 ----
    float* Cst = Wlds;                                // reuse W region as C staging
    {
        const int wave   = t >> 5;                    // 8 waves -> 8 pixels
        const int lane   = t & 31;
        const int laneLo = lane & 15;
        const int hiHalf = lane >> 4;                 // 0: lanes 0-15, 1: lanes 16-31
        const int p      = wave;

        // B fragment: column = laneLo; lanes0-15: K0-7 & K16-23; lanes16-31: K8-15 & K24-31
        AFrag bf;
        const _Float16* bcol = &S16[(p*Bb + laneLo)*Nn + hiHalf*8];
        bf.h[0] = *(const v8h*)(bcol);
        bf.h[1] = *(const v8h*)(bcol + 16);

        #pragma unroll
        for (int rb = 0; rb < 2; ++rb) {
            // A: row = rb*16+laneLo; lanes0-15: K0-7 & K16-23; lanes16-31: K8-15 & K24-31
            const int row = rb*16 + laneLo;
            const _Float16* arow = &A16[(p*Nn + row)*Nn + hiHalf*8];
            AFrag af;
            af.h[0] = *(const v8h*)(arow);
            af.h[1] = *(const v8h*)(arow + 16);

            CFrag cf;
            #pragma unroll
            for (int r = 0; r < 8; ++r) cf.f[r] = 0.0f;
            cf.v = __builtin_amdgcn_wmma_f32_16x16x32_f16(
                       false, af.v, false, bf.v, (short)0, cf.v, false, false);

            // C layout: VGPR r -> M = r + 8*hiHalf ; N = laneLo
            const int mbase = rb*16 + hiHalf*8;
            #pragma unroll
            for (int r = 0; r < 8; ++r)
                Cst[(p*Nn + (mbase + r))*Bb + laneLo] = cf.f[r];
        }
    }
    __syncthreads();

    // ---- Phase 4: writeback as b128 (gather 4 pixels per lane from LDS) ----
    #pragma unroll
    for (int k = 0; k < (PIX*Nn*Bb/4)/256; ++k) {
        int idx = t + k*256;
        int p4 = idx & 1, rest = idx >> 1;
        int b = rest & 15, n = rest >> 4;
        int pb = p4*4;
        float4 v;
        v.x = Cst[((pb+0)*Nn + n)*Bb + b];
        v.y = Cst[((pb+1)*Nn + n)*Bb + b];
        v.z = Cst[((pb+2)*Nn + n)*Bb + b];
        v.w = Cst[((pb+3)*Nn + n)*Bb + b];
        *(float4*)(out + (n*Bb + b)*HWp + pix0 + pb) = v;
    }
}

extern "C" void kernel_launch(void* const* d_in, const int* in_sizes, int n_in,
                              void* d_out, int out_size, void* d_ws, size_t ws_size,
                              hipStream_t stream) {
    const float* spikes = (const float*)d_in[0];
    const float* exc    = (const float*)d_in[1];
    const float* cw     = (const float*)d_in[2];
    const float* sw     = (const float*)d_in[3];
    const int*   cs     = (const int*)d_in[4];
    const int*   cd     = (const int*)d_in[5];
    float*       out    = (float*)d_out;
    (void)in_sizes; (void)n_in; (void)out_size; (void)d_ws; (void)ws_size;

    dim3 grid(HWp / PIX);   // 2048 workgroups, 8 pixels each
    scs_wmma_kernel<<<grid, 256, 0, stream>>>(spikes, exc, cw, sw, cs, cd, out);
}
